// Transducer_60129542144503
// MI455X (gfx1250) — matrix operand order repaired and checked by reading, not compiled, not run on hardware
//
#include <hip/hip_runtime.h>
#include <cstddef>

// Problem constants (match reference)
#define Bb   8
#define Tt   200
#define FIN  80
#define Uu   100
#define UP1  101      // U+1
#define Hh   320
#define Vv   128
#define G4H  (4*Hh)   // 1280
#define TUCNT (Tt*UP1) // 20200

typedef __attribute__((ext_vector_type(16))) _Float16 v16h;
typedef __attribute__((ext_vector_type(8)))  float    v8f;

#define WMMA_F16(a, b, c) \
  __builtin_amdgcn_wmma_f32_16x16x32_f16(false, (a), false, (b), (short)0, (c), false, false)

// ---------------------------------------------------------------- helpers
__device__ __forceinline__ float sigmoidf(float x) { return 1.0f / (1.0f + __expf(-x)); }

__device__ __forceinline__ float logaddexpf(float a, float b) {
  float mx = fmaxf(a, b);
  float mn = fminf(a, b);
  if (mx == -INFINITY) return -INFINITY;
  return mx + log1pf(__expf(mn - mx));
}

// A-fragment (16x32 f16, M x K) from global row-major A[M][K].
// ISA 7.12.2 16-bit A layout: lanes 0-15 row=lane, elems 0..7 = K k0..k0+7,
// elems 8..15 = K k0+16..k0+23; lanes 16-31 same rows, +8 K offset.
// GUARD_M: rows >= M return zeros (only needed when M % 16 != 0).
template <bool GUARD_M>
__device__ __forceinline__ v16h load_a_frag_g(const _Float16* __restrict__ A,
                                              int M, int K, int mBase, int k0, int lane) {
  v16h a;
  int r  = mBase + (lane & 15);
  int kb = k0 + ((lane & 16) ? 8 : 0);
  if (!GUARD_M || r < M) {
    const _Float16* p = A + (size_t)r * K + kb;
#pragma unroll
    for (int e = 0; e < 8; ++e) { a[e] = p[e]; a[8 + e] = p[16 + e]; }
  } else {
#pragma unroll
    for (int e = 0; e < 16; ++e) a[e] = (_Float16)0.0f;
  }
  return a;
}

// Same A-fragment but from an LDS tile with row stride (in halves) = zstride.
__device__ __forceinline__ v16h load_a_frag_lds(const _Float16* __restrict__ zsh,
                                                int zstride, int k0, int lane) {
  v16h a;
  int r  = lane & 15;
  int kb = k0 + ((lane & 16) ? 8 : 0);
  const _Float16* p = zsh + r * zstride + kb;
#pragma unroll
  for (int e = 0; e < 8; ++e) { a[e] = p[e]; a[8 + e] = p[16 + e]; }
  return a;
}

// B-fragment (32x16 f16, K x N) where B = W^T and W is row-major W[N][K].
// Same K order as the A fragment: lanes 0-15 col = nBase+lane, elems 0..7 =
// K k0..k0+7, elems 8..15 = K k0+16..k0+23; lanes 16-31 same cols, +8 K offset.
__device__ __forceinline__ v16h load_b_frag(const _Float16* __restrict__ W,
                                            int K, int nBase, int k0, int lane) {
  int n  = nBase + (lane & 15);
  int kb = k0 + ((lane & 16) ? 8 : 0);
  const _Float16* p = W + (size_t)n * K + kb;
  v16h b;
#pragma unroll
  for (int e = 0; e < 8; ++e) { b[e] = p[e]; b[8 + e] = p[16 + e]; }
  return b;
}

// ---------------------------------------------------------------- kernels

// x0[b,t,f] = (xs[b,0,f,t]*cw1 + cb1)*cw2 + cb2
__global__ void prep_x_kernel(const float* __restrict__ xs,
                              const float* __restrict__ cw1, const float* __restrict__ cb1,
                              const float* __restrict__ cw2, const float* __restrict__ cb2,
                              float* __restrict__ x0) {
  int idx = blockIdx.x * blockDim.x + threadIdx.x;
  int n = Bb * Tt * FIN;
  if (idx >= n) return;
  int f = idx % FIN;
  int t = (idx / FIN) % Tt;
  int b = idx / (FIN * Tt);
  float v = xs[((size_t)b * FIN + f) * Tt + t];
  v = (v * cw1[0] + cb1[0]) * cw2[0] + cb2[0];
  x0[idx] = v;
}

// g_in[b,u,:] = embed[y0[b,u]], y0 = [0, ys...]; embed row length = V-1 = 127
__global__ void embed_kernel(const int* __restrict__ ys,
                             const float* __restrict__ embed,
                             float* __restrict__ g_in) {
  int idx = blockIdx.x * blockDim.x + threadIdx.x;
  const int E = Vv - 1;
  int n = Bb * UP1 * E;
  if (idx >= n) return;
  int k = idx % E;
  int u = (idx / E) % UP1;
  int b = idx / (E * UP1);
  int tok = (u == 0) ? 0 : ys[b * Uu + (u - 1)];
  g_in[idx] = embed[(size_t)tok * E + k];
}

__global__ void cvt_f16_kernel(const float* __restrict__ src, _Float16* __restrict__ dst, int n) {
  int i = blockIdx.x * blockDim.x + threadIdx.x;
  if (i < n) dst[i] = (_Float16)src[i];
}

// Strided 2D convert (slices Wf/Wg out of jW1 (H x 2H))
__global__ void cvt_f16_strided_kernel(const float* __restrict__ src, int srcStride,
                                       _Float16* __restrict__ dst, int rows, int cols) {
  int i = blockIdx.x * blockDim.x + threadIdx.x;
  if (i >= rows * cols) return;
  int c = i % cols, r = i / cols;
  dst[i] = (_Float16)src[(size_t)r * srcStride + c];
}

// Convert rows x scols f32 -> rows x dcols f16, zero-padding cols >= scols
// (pads K of the input/Wih matrices up to a multiple of 32 for WMMA)
__global__ void cvt_pad_f16_kernel(const float* __restrict__ src, int scols,
                                   _Float16* __restrict__ dst, int dcols, int rows) {
  int i = blockIdx.x * blockDim.x + threadIdx.x;
  if (i >= rows * dcols) return;
  int c = i % dcols, r = i / dcols;
  dst[i] = (c < scols) ? (_Float16)src[(size_t)r * scols + c] : (_Float16)0.0f;
}

// bsum = bih + bhh  (folded into the gatesX GEMM bias)
__global__ void bias_sum_kernel(const float* __restrict__ a, const float* __restrict__ b,
                                float* __restrict__ o, int n) {
  int i = blockIdx.x * blockDim.x + threadIdx.x;
  if (i < n) o[i] = a[i] + b[i];
}

// C[M,N] = A[M,K] @ W[N,K]^T (+ bias[N]);  f16 in, f32 out.
// One wave per 16x32 strip of C: the A fragment is loaded once per k-step and
// feeds two independent WMMAs (operand reuse + co-issue instead of hazard NOPs).
// grid = (N/32, ceil(M/16)), block = 32.  N must be a multiple of 32.
template <bool GUARD_M>
__global__ __launch_bounds__(32)
void gemm_wmma_kernel(const _Float16* __restrict__ A, const _Float16* __restrict__ W,
                      const float* __restrict__ bias, float* __restrict__ C,
                      int M, int N, int K) {
  int lane = threadIdx.x;
  int nBase0 = blockIdx.x * 32;
  int nBase1 = nBase0 + 16;
  int mBase = blockIdx.y * 16;
  v8f acc0 = {0.f, 0.f, 0.f, 0.f, 0.f, 0.f, 0.f, 0.f};
  v8f acc1 = {0.f, 0.f, 0.f, 0.f, 0.f, 0.f, 0.f, 0.f};
  for (int k0 = 0; k0 < K; k0 += 32) {
    v16h a  = load_a_frag_g<GUARD_M>(A, M, K, mBase, k0, lane);
    v16h b0 = load_b_frag(W, K, nBase0, k0, lane);
    v16h b1 = load_b_frag(W, K, nBase1, k0, lane);
    acc0 = WMMA_F16(a, b0, acc0);
    acc1 = WMMA_F16(a, b1, acc1);
  }
  int nl = lane & 15;
  int rb = (lane & 16) ? 8 : 0;
  float bi0 = bias ? bias[nBase0 + nl] : 0.0f;
  float bi1 = bias ? bias[nBase1 + nl] : 0.0f;
#pragma unroll
  for (int r = 0; r < 8; ++r) {
    int m = mBase + rb + r;
    if (!GUARD_M || m < M) {
      C[(size_t)m * N + nBase0 + nl] = acc0[r] + bi0;
      C[(size_t)m * N + nBase1 + nl] = acc1[r] + bi1;
    }
  }
}

// ------------------------------------------------------------------
// Recurrent half of an LSTM layer, WMMA-based.
// gatesX (B,Tlen,4H) f32 already holds x_t@Wih^T + bih + bhh.
// Per step:  gates = gatesX[:,t,:] + h_{t-1} @ Whh^T   (M=16 rows, 8 valid)
// h tile lives in LDS as f16 (rows 8..15 stay zero); 8 waves cover the
// 80 N-tiles of 4H=1280 in pairs (shared A fragment); K=320 -> 10 k-steps.
// hs is written as f16 because every consumer is an f16 WMMA GEMM.
#define HST 328   // LDS h-tile row stride in halves (padded vs 320)
__global__ __launch_bounds__(256)
void lstm_rec_wmma_kernel(const float* __restrict__ gatesX,   // (B,Tlen,4H)
                          const _Float16* __restrict__ Whh16, // (4H,H)
                          _Float16* __restrict__ hs,          // (B,Tlen,H) f16
                          int Tlen) {
  __shared__ _Float16 h16[16 * HST];      // ~10.5 KB (rows 8..15 zero)
  __shared__ float    gates_s[Bb * G4H];  // 40 KB
  __shared__ float    c_s[Bb * Hh];       // 10 KB
  int tid = threadIdx.x;
  int lane = tid & 31, wave = tid >> 5;
  for (int i = tid; i < 16 * HST; i += 256) h16[i] = (_Float16)0.0f;
  for (int i = tid; i < Bb * Hh; i += 256) c_s[i] = 0.0f;
  __syncthreads();
  for (int t = 0; t < Tlen; ++t) {
    // gates_s[b][n] = h_{t-1} @ Whh^T : tiles walked in pairs (nt, nt+8)
    for (int nt = wave; nt < G4H / 16; nt += 16) {
      int nBase0 = nt * 16;
      int nBase1 = (nt + 8) * 16;
      v8f acc0 = {0.f, 0.f, 0.f, 0.f, 0.f, 0.f, 0.f, 0.f};
      v8f acc1 = {0.f, 0.f, 0.f, 0.f, 0.f, 0.f, 0.f, 0.f};
      for (int k0 = 0; k0 < Hh; k0 += 32) {
        v16h a  = load_a_frag_lds(h16, HST, k0, lane);
        v16h b0 = load_b_frag(Whh16, Hh, nBase0, k0, lane);
        v16h b1 = load_b_frag(Whh16, Hh, nBase1, k0, lane);
        acc0 = WMMA_F16(a, b0, acc0);
        acc1 = WMMA_F16(a, b1, acc1);
      }
      if (!(lane & 16)) {              // C rows 0..7 (the batch rows) live in lanes 0..15
#pragma unroll
        for (int r = 0; r < 8; ++r) {
          gates_s[r * G4H + nBase0 + lane] = acc0[r];
          gates_s[r * G4H + nBase1 + lane] = acc1[r];
        }
      }
    }
    __syncthreads();
    // c = sig(f)*c + sig(i)*tanh(g); h = sig(o)*tanh(c)
    for (int idx = tid; idx < Bb * Hh; idx += 256) {
      int b = idx / Hh, j = idx % Hh;
      const float* gx = gatesX + ((size_t)b * Tlen + t) * G4H;
      const float* gs = gates_s + b * G4H;
      float gi = gx[j]          + gs[j];
      float gf = gx[Hh + j]     + gs[Hh + j];
      float gg = gx[2 * Hh + j] + gs[2 * Hh + j];
      float go = gx[3 * Hh + j] + gs[3 * Hh + j];
      float c = sigmoidf(gf) * c_s[idx] + sigmoidf(gi) * tanhf(gg);
      float h = sigmoidf(go) * tanhf(c);
      c_s[idx] = c;
      _Float16 hv = (_Float16)h;
      h16[b * HST + j] = hv;
      hs[((size_t)b * Tlen + t) * Hh + j] = hv;
    }
    __syncthreads();
  }
}

// Fused joint network: for 16 (t,u) cells of one batch element:
//   z = tanh(fproj[t]+gproj[u]+jb1) (f16, LDS)  ->  out = z @ jW2^T + jb2 (WMMA)
//   -> log-softmax over V=128 -> store blank_lp and emit_lp(label) only.
#define ZST 328   // LDS row stride in halves
__global__ __launch_bounds__(128)
void joint_kernel(const float* __restrict__ fproj,   // (B,T,H)
                  const float* __restrict__ gproj,   // (B,U+1,H)
                  const float* __restrict__ jb1,     // (H)
                  const _Float16* __restrict__ jW2,  // (V,H) f16
                  const float* __restrict__ jb2,     // (V)
                  const int* __restrict__ ys,        // (B,U)
                  float* __restrict__ blank_lp,      // (B,T,U+1)
                  float* __restrict__ emit_lp) {     // (B,T,U)
  __shared__ _Float16 zsh[16 * ZST];   // ~10.5 KB
  __shared__ float outsh[16 * Vv];     // 8 KB
  int tid = threadIdx.x;
  int b = blockIdx.y;
  int tuBase = blockIdx.x * 16;

  // Phase 1: build 16 rows of z as f16
  for (int idx = tid; idx < 16 * Hh; idx += 128) {
    int m = idx / Hh, hh = idx % Hh;
    int tu = tuBase + m;
    float v = 0.0f;
    if (tu < TUCNT) {
      int t = tu / UP1, u = tu % UP1;
      v = tanhf(fproj[((size_t)b * Tt + t) * Hh + hh] +
                gproj[((size_t)b * UP1 + u) * Hh + hh] + jb1[hh]);
    }
    zsh[m * ZST + hh] = (_Float16)v;
  }
  __syncthreads();

  // Phase 2: 8 WMMA N-tiles (V=128), 4 waves x 2 tiles with shared A fragment
  int lane = tid & 31;
  int wave = tid >> 5;
  {
    int nBase0 = wave * 16;
    int nBase1 = nBase0 + 64;   // second half of V
    v8f acc0 = {0.f, 0.f, 0.f, 0.f, 0.f, 0.f, 0.f, 0.f};
    v8f acc1 = {0.f, 0.f, 0.f, 0.f, 0.f, 0.f, 0.f, 0.f};
    for (int k0 = 0; k0 < Hh; k0 += 32) {
      v16h a  = load_a_frag_lds(zsh, ZST, k0, lane);
      v16h b0 = load_b_frag(jW2, Hh, nBase0, k0, lane);
      v16h b1 = load_b_frag(jW2, Hh, nBase1, k0, lane);
      acc0 = WMMA_F16(a, b0, acc0);
      acc1 = WMMA_F16(a, b1, acc1);
    }
    int nl = lane & 15;
    int rb = (lane & 16) ? 8 : 0;
    float bi0 = jb2[nBase0 + nl];
    float bi1 = jb2[nBase1 + nl];
#pragma unroll
    for (int r = 0; r < 8; ++r) {
      outsh[(rb + r) * Vv + nBase0 + nl] = acc0[r] + bi0;
      outsh[(rb + r) * Vv + nBase1 + nl] = acc1[r] + bi1;
    }
  }
  __syncthreads();

  // Phase 3: log-softmax per row; keep only blank and label log-probs
  if (tid < 16) {
    int tu = tuBase + tid;
    if (tu < TUCNT) {
      const float* row = outsh + tid * Vv;
      float mx = -INFINITY;
      for (int v = 0; v < Vv; ++v) mx = fmaxf(mx, row[v]);
      float s = 0.0f;
      for (int v = 0; v < Vv; ++v) s += __expf(row[v] - mx);
      float lse = mx + __logf(s);
      int t = tu / UP1, u = tu % UP1;
      blank_lp[((size_t)b * Tt + t) * UP1 + u] = row[0] - lse;
      if (u < Uu) {
        int lbl = ys[b * Uu + u];
        emit_lp[((size_t)b * Tt + t) * Uu + u] = row[lbl] - lse;
      }
    }
  }
}

// RNN-T forward DP in log space; one lane per batch element; scalar loss out.
__global__ __launch_bounds__(32)
void rnnt_loss_kernel(const float* __restrict__ blank_lp,
                      const float* __restrict__ emit_lp,
                      float* __restrict__ out) {
  __shared__ float alpha[Bb][UP1];
  __shared__ float llsh[Bb];
  int tid = threadIdx.x;
  if (tid < Bb) {
    int b = tid;
    float* a = alpha[b];
    a[0] = 0.0f;
    for (int u = 1; u <= Uu; ++u)
      a[u] = a[u - 1] + emit_lp[((size_t)b * Tt + 0) * Uu + (u - 1)];
    for (int t = 1; t < Tt; ++t) {
      const float* bl = blank_lp + ((size_t)b * Tt + (t - 1)) * UP1;
      const float* em = emit_lp + ((size_t)b * Tt + t) * Uu;
      float c = a[0] + bl[0];
      a[0] = c;
      for (int u = 1; u <= Uu; ++u) {
        float au = a[u] + bl[u];
        c = logaddexpf(au, c + em[u - 1]);
        a[u] = c;
      }
    }
    llsh[b] = a[Uu] + blank_lp[((size_t)b * Tt + (Tt - 1)) * UP1 + Uu];
  }
  __syncthreads();
  if (tid == 0) {
    float s = 0.0f;
    for (int b = 0; b < Bb; ++b) s += llsh[b];
    out[0] = -(s / (float)Bb);
  }
}

// ---------------------------------------------------------------- launch
extern "C" void kernel_launch(void* const* d_in, const int* in_sizes, int n_in,
                              void* d_out, int out_size, void* d_ws, size_t ws_size,
                              hipStream_t stream) {
  // Input order: top-level dict insertion order {xs, ys, xlen, ylen, params, embed};
  // params flattened in JAX pytree (sorted-key) order:
  //   cb1,cb2,cw1,cw2, dec_lstm(Wih,Whh,bih,bhh), enc_lstm[2](Wih,Whh,bih,bhh),
  //   encW,encb, jW1,jW2, jb1,jb2
  const float* xs     = (const float*)d_in[0];
  const int*   ys     = (const int*)  d_in[1];
  const float* cb1    = (const float*)d_in[4];
  const float* cb2    = (const float*)d_in[5];
  const float* cw1    = (const float*)d_in[6];
  const float* cw2    = (const float*)d_in[7];
  const float* dWih   = (const float*)d_in[8];
  const float* dWhh   = (const float*)d_in[9];
  const float* dbih   = (const float*)d_in[10];
  const float* dbhh   = (const float*)d_in[11];
  const float* e0Wih  = (const float*)d_in[12];
  const float* e0Whh  = (const float*)d_in[13];
  const float* e0bih  = (const float*)d_in[14];
  const float* e0bhh  = (const float*)d_in[15];
  const float* e1Wih  = (const float*)d_in[16];
  const float* e1Whh  = (const float*)d_in[17];
  const float* e1bih  = (const float*)d_in[18];
  const float* e1bhh  = (const float*)d_in[19];
  const float* encW   = (const float*)d_in[20];
  const float* encb   = (const float*)d_in[21];
  const float* jW1    = (const float*)d_in[22];
  const float* jW2    = (const float*)d_in[23];
  const float* jb1    = (const float*)d_in[24];
  const float* jb2    = (const float*)d_in[25];
  const float* embed  = (const float*)d_in[26];
  float* outp = (float*)d_out;

  // Carve workspace (total ~25 MB)
  char* wp = (char*)d_ws;
  auto carve = [&](size_t bytes) -> void* {
    void* p = (void*)wp;
    wp += (bytes + 255) & ~(size_t)255;
    return p;
  };
  const int MT  = Bb * Tt;    // 1600  (multiple of 16 -> unguarded GEMMs)
  const int MU  = Bb * UP1;   // 808   (not a multiple of 16 -> guarded GEMMs)
  const int KF  = 96;         // FIN=80 padded to 32 multiple
  const int KE  = 128;        // V-1=127 padded to 32 multiple
  float*    x0       = (float*)   carve((size_t)MT * FIN * 4);
  float*    g_in     = (float*)   carve((size_t)MU * (Vv - 1) * 4);
  float*    gatesX   = (float*)   carve((size_t)MT * G4H * 4);   // reused for all 3 LSTMs
  _Float16* x0p16    = (_Float16*)carve((size_t)MT * KF * 2);
  _Float16* g_inp16  = (_Float16*)carve((size_t)MU * KE * 2);
  _Float16* h1_16    = (_Float16*)carve((size_t)MT * Hh * 2);
  _Float16* h2_16    = (_Float16*)carve((size_t)MT * Hh * 2);
  _Float16* g16      = (_Float16*)carve((size_t)MU * Hh * 2);
  float*    f32f     = (float*)   carve((size_t)MT * Hh * 4);
  _Float16* f16f     = (_Float16*)carve((size_t)MT * Hh * 2);
  float*    fproj    = (float*)   carve((size_t)MT * Hh * 4);
  float*    gproj    = (float*)   carve((size_t)MU * Hh * 4);
  _Float16* e0Wih16  = (_Float16*)carve((size_t)G4H * KF * 2);
  _Float16* e1Wih16  = (_Float16*)carve((size_t)G4H * Hh * 2);
  _Float16* dWih16   = (_Float16*)carve((size_t)G4H * KE * 2);
  _Float16* e0Whh16  = (_Float16*)carve((size_t)G4H * Hh * 2);
  _Float16* e1Whh16  = (_Float16*)carve((size_t)G4H * Hh * 2);
  _Float16* dWhh16   = (_Float16*)carve((size_t)G4H * Hh * 2);
  _Float16* encW16   = (_Float16*)carve((size_t)Hh * Hh * 2);
  _Float16* Wf16     = (_Float16*)carve((size_t)Hh * Hh * 2);
  _Float16* Wg16     = (_Float16*)carve((size_t)Hh * Hh * 2);
  _Float16* jW2_16   = (_Float16*)carve((size_t)Vv * Hh * 2);
  float*    bs0      = (float*)   carve((size_t)G4H * 4);
  float*    bs1      = (float*)   carve((size_t)G4H * 4);
  float*    bsd      = (float*)   carve((size_t)G4H * 4);
  float*    blank_lp = (float*)   carve((size_t)Bb * Tt * UP1 * 4);
  float*    emit_lp  = (float*)   carve((size_t)Bb * Tt * Uu * 4);
  (void)ws_size; (void)in_sizes; (void)n_in; (void)out_size;

  auto blocks = [](int n) { return (n + 255) / 256; };

  // ---- weight prep (f16, K padded where needed) + bias folding
  cvt_pad_f16_kernel<<<blocks(G4H * KF), 256, 0, stream>>>(e0Wih, FIN, e0Wih16, KF, G4H);
  cvt_f16_kernel<<<blocks(G4H * Hh), 256, 0, stream>>>(e1Wih, e1Wih16, G4H * Hh);
  cvt_pad_f16_kernel<<<blocks(G4H * KE), 256, 0, stream>>>(dWih, Vv - 1, dWih16, KE, G4H);
  cvt_f16_kernel<<<blocks(G4H * Hh), 256, 0, stream>>>(e0Whh, e0Whh16, G4H * Hh);
  cvt_f16_kernel<<<blocks(G4H * Hh), 256, 0, stream>>>(e1Whh, e1Whh16, G4H * Hh);
  cvt_f16_kernel<<<blocks(G4H * Hh), 256, 0, stream>>>(dWhh, dWhh16, G4H * Hh);
  cvt_f16_kernel<<<blocks(Hh * Hh), 256, 0, stream>>>(encW, encW16, Hh * Hh);
  cvt_f16_strided_kernel<<<blocks(Hh * Hh), 256, 0, stream>>>(jW1, 2 * Hh, Wf16, Hh, Hh);
  cvt_f16_strided_kernel<<<blocks(Hh * Hh), 256, 0, stream>>>(jW1 + Hh, 2 * Hh, Wg16, Hh, Hh);
  cvt_f16_kernel<<<blocks(Vv * Hh), 256, 0, stream>>>(jW2, jW2_16, Vv * Hh);
  bias_sum_kernel<<<blocks(G4H), 256, 0, stream>>>(e0bih, e0bhh, bs0, G4H);
  bias_sum_kernel<<<blocks(G4H), 256, 0, stream>>>(e1bih, e1bhh, bs1, G4H);
  bias_sum_kernel<<<blocks(G4H), 256, 0, stream>>>(dbih, dbhh, bsd, G4H);

  // ---- inputs
  prep_x_kernel<<<blocks(MT * FIN), 256, 0, stream>>>(xs, cw1, cb1, cw2, cb2, x0);
  cvt_pad_f16_kernel<<<blocks(MT * KF), 256, 0, stream>>>(x0, FIN, x0p16, KF, MT);
  embed_kernel<<<blocks(MU * (Vv - 1)), 256, 0, stream>>>(ys, embed, g_in);
  cvt_pad_f16_kernel<<<blocks(MU * KE), 256, 0, stream>>>(g_in, Vv - 1, g_inp16, KE, MU);

  // ---- encoder LSTM 0: gatesX GEMM (parallel) + WMMA recurrence (serial)
  gemm_wmma_kernel<false><<<dim3(G4H / 32, MT / 16), 32, 0, stream>>>(
      x0p16, e0Wih16, bs0, gatesX, MT, G4H, KF);
  lstm_rec_wmma_kernel<<<1, 256, 0, stream>>>(gatesX, e0Whh16, h1_16, Tt);
  // ---- encoder LSTM 1
  gemm_wmma_kernel<false><<<dim3(G4H / 32, MT / 16), 32, 0, stream>>>(
      h1_16, e1Wih16, bs1, gatesX, MT, G4H, Hh);
  lstm_rec_wmma_kernel<<<1, 256, 0, stream>>>(gatesX, e1Whh16, h2_16, Tt);
  // ---- decoder LSTM
  gemm_wmma_kernel<true><<<dim3(G4H / 32, (MU + 15) / 16), 32, 0, stream>>>(
      g_inp16, dWih16, bsd, gatesX, MU, G4H, KE);
  lstm_rec_wmma_kernel<<<1, 256, 0, stream>>>(gatesX, dWhh16, g16, UP1);

  // ---- f = h2 @ encW^T + encb ; fproj = f @ Wf^T ; gproj = g @ Wg^T
  gemm_wmma_kernel<false><<<dim3(Hh / 32, MT / 16), 32, 0, stream>>>(
      h2_16, encW16, encb, f32f, MT, Hh, Hh);
  cvt_f16_kernel<<<blocks(MT * Hh), 256, 0, stream>>>(f32f, f16f, MT * Hh);
  gemm_wmma_kernel<false><<<dim3(Hh / 32, MT / 16), 32, 0, stream>>>(
      f16f, Wf16, nullptr, fproj, MT, Hh, Hh);
  gemm_wmma_kernel<true><<<dim3(Hh / 32, (MU + 15) / 16), 32, 0, stream>>>(
      g16, Wg16, nullptr, gproj, MU, Hh, Hh);

  // ---- fused joint: tanh-combine -> WMMA x jW2 -> log-softmax -> gather
  joint_kernel<<<dim3((TUCNT + 15) / 16, Bb), 128, 0, stream>>>(
      fproj, gproj, jb1, jW2_16, jb2, ys, blank_lp, emit_lp);
  // ---- RNNT DP + scalar loss
  rnnt_loss_kernel<<<1, 32, 0, stream>>>(blank_lp, emit_lp, outp);
}
